// TtMultiHeadAttentionModel_9921374454152
// MI455X (gfx1250) — matrix operand order repaired and checked by reading, not compiled, not run on hardware
//
#include <hip/hip_runtime.h>

// ---------------------------------------------------------------------------
// MHA for MI455X (gfx1250, wave32): bf16 WMMA + TDM (tensor_load_to_lds)
// B=8, S=1024, H=1024, heads=16, d=64
// ---------------------------------------------------------------------------

typedef __attribute__((ext_vector_type(16))) __bf16 v16bf;
typedef __attribute__((ext_vector_type(8)))  __bf16 v8bf;
typedef __attribute__((ext_vector_type(8)))  float  v8f;
typedef __attribute__((ext_vector_type(4)))  unsigned v4u;
typedef __attribute__((ext_vector_type(8)))  int    v8i;
typedef __attribute__((ext_vector_type(4)))  int    v4i;

#define HDIM 1024
#define SEQ  1024
#define NB   8
#define NH   16
#define DH   64
#define MROWS (NB * SEQ)       // 8192

#define KSTEP   64
#define NSTAGE  (HDIM / KSTEP) // 16
#define LDPAD   72             // 64 elems + 8-elem (16B) pad per row
#define XTILE_E (128 * LDPAD)  // 9216 elems = 18432 B
#define WTILE_E (64 * LDPAD)   //  4608 elems =  9216 B

#if __has_builtin(__builtin_amdgcn_tensor_load_to_lds)
#define USE_TDM 1
#else
#define USE_TDM 0
#endif

// The TDM writes LDS behind the compiler's back (the builtin has no pointer
// argument). A bare "memory"-clobber asm is NOT enough: GlobalOpt only looks
// at users of the LDS *symbol*, sees no stores and no captures, and folds all
// loads to undef. Passing a pointer derived from the LDS array INTO the asm
// captures the global (GlobalOpt bails) and the memory clobber forces each
// stage to genuinely re-read LDS.
static __device__ __forceinline__ void lds_escape(const void* p) {
  asm volatile("" :: "v"(p) : "memory");
}

// ---- helpers --------------------------------------------------------------

static __device__ __forceinline__ __bf16 f2bf(float f) {
  union { float f; unsigned u; } a; a.f = f;
  unsigned r = a.u + 0x7FFFu + ((a.u >> 16) & 1u);   // round-to-nearest-even
  union { unsigned short s; __bf16 b; } o;
  o.s = (unsigned short)(r >> 16);
  return o.b;
}

static __device__ __forceinline__ v8f vzero8() {
  v8f z;
  for (int i = 0; i < 8; ++i) z[i] = 0.0f;
  return z;
}

// A-fragment (16x32 bf16, MxK): lane = M | (khalf<<4); per lane two 16B
// chunks at (kb) and (kb+16), kb = k0 + (lane>=16 ? 8 : 0).
static __device__ __forceinline__ v16bf load_a_frag(const __bf16* base, int ld,
                                                    int lane, int k0) {
  int r  = lane & 15;
  int kb = k0 + ((lane >> 4) << 3);
  const __bf16* p = base + (size_t)r * ld + kb;
  union { v16bf v; v8bf h[2]; } u;
  u.h[0] = *(const v8bf*)(p);
  u.h[1] = *(const v8bf*)(p + 16);
  return u.v;
}

// B-fragment (32x16 bf16, KxN) from row-major B^T (N rows, K contiguous):
// lane = N | (khalf<<4); per lane two 16B chunks at (kb) and (kb+16),
// kb = k0 + (lane>=16 ? 8 : 0).
static __device__ __forceinline__ v16bf load_b_frag(const __bf16* base, int ld,
                                                    int lane, int k0) {
  int n  = lane & 15;
  int kb = k0 + ((lane >> 4) << 3);
  const __bf16* p = base + (size_t)n * ld + kb;
  union { v16bf v; v8bf h[2]; } u;
  u.h[0] = *(const v8bf*)(p);
  u.h[1] = *(const v8bf*)(p + 16);
  return u.v;
}

static __device__ __forceinline__ v8f wmma_bf16(v16bf a, v16bf b, v8f c) {
  return __builtin_amdgcn_wmma_f32_16x16x32_bf16(
      false, a, false, b, (short)0, c, false, false);
}

// ---- TDM 2D tile load: global (row-major, ld = row_len elems of bf16) -> LDS
// with a 16B pad inserted after every 128B row (pad_interval=32 dw, pad=4 dw).

#if USE_TDM
static __device__ __forceinline__ void tdm_load_2d(const __bf16* gbase,
                                                   unsigned lds_byte_off,
                                                   int tile_cols, int tile_rows,
                                                   int row_len, int nrows) {
  unsigned long long ga = (unsigned long long)(size_t)gbase;
  v4u g0;
  g0[0] = 1u;                                   // count=1, user descriptor
  g0[1] = lds_byte_off;                         // lds_addr
  g0[2] = (unsigned)(ga & 0xFFFFFFFFu);         // global_addr[31:0]
  g0[3] = (unsigned)((ga >> 32) & 0x01FFFFFFu)  // global_addr[56:32]
          | (2u << 30);                          // type=2 ("image")
  v8i g1;
  g1[0] = (int)((1u << 16)        // data_size = 2 bytes
              | (1u << 20)        // pad_enable
              | (4u << 22)        // pad_interval: 32 DWORDs (=128B row)
              | (3u << 25));      // pad_amount:   4 DWORDs (=16B)
  g1[1] = (int)(((unsigned)row_len & 0xFFFFu) << 16);            // tensor_dim0 lo
  g1[2] = (int)(((unsigned)row_len >> 16) |
                (((unsigned)nrows & 0xFFFFu) << 16));            // dim0 hi | dim1 lo
  g1[3] = (int)(((unsigned)nrows >> 16) |
                ((unsigned)tile_cols << 16));                    // dim1 hi | tile_dim0
  g1[4] = (int)(unsigned)tile_rows;                              // tile_dim1, tile_dim2=0
  g1[5] = (int)(unsigned)row_len;                                // tensor_dim0_stride lo
  g1[6] = 0;
  g1[7] = 0;
  v4i z4 = {0, 0, 0, 0};
#if defined(__clang_major__) && (__clang_major__ >= 23)
  v8i z8 = {0, 0, 0, 0, 0, 0, 0, 0};
  __builtin_amdgcn_tensor_load_to_lds(g0, g1, z4, z4, z8, 0);
#else
  __builtin_amdgcn_tensor_load_to_lds(g0, g1, z4, z4, 0);
#endif
}
#endif

// ---- kernel 1: f32 -> bf16 ------------------------------------------------

__global__ void cvt_bf16(const float* __restrict__ in, __bf16* __restrict__ out,
                         int n) {
  int i = blockIdx.x * blockDim.x + threadIdx.x;
  int stride = gridDim.x * blockDim.x;
  for (; i < n; i += stride) out[i] = f2bf(in[i]);
}

// ---- kernel 2: Y = X * W^T + b --------------------------------------------
// Block = 256 threads (8 waves, 4M x 2N), macro-tile 128x64, K-stages of 64,
// double-buffered LDS filled by TDM (wave 0 drives the DMA).
// vmode 0: Y[(b*NH+h)*SEQ*DH + s*DH + dc]   (Q, K: (b,h,s,d))
// vmode 1: Y[(b*NH+h)*SEQ*DH + dc*SEQ + s]  (V stored transposed: (b,h,d,s))

static __device__ __forceinline__ void store_tile_bf16(
    __bf16* __restrict__ Y, const float* __restrict__ bias, v8f acc,
    int gm0, int gn0, int vmode, int lane) {
  int n  = gn0 + (lane & 15);
  int h  = n >> 6;
  int dc = n & 63;
  float bv = bias[n];
  int mofs = (lane >> 4) << 3;
  for (int r = 0; r < 8; ++r) {
    int m = gm0 + mofs + r;
    int b = m >> 10;
    int s = m & (SEQ - 1);
    size_t head = (size_t)(b * NH + h) * (SEQ * DH);
    size_t idx  = vmode ? head + (size_t)dc * SEQ + s
                        : head + (size_t)s * DH + dc;
    Y[idx] = f2bf(acc[r] + bv);
  }
}

static __device__ __forceinline__ void fetch_stage(
    const __bf16* Xg, const __bf16* Wg, int k0,
    __bf16* xs, __bf16* ws, unsigned xs_off, unsigned ws_off, int tid) {
#if USE_TDM
  (void)xs; (void)ws;
  if ((tid >> 5) == 0) {   // wave 0 drives the tensor DMA (EXEC ignored by TDM)
    tdm_load_2d(Xg + k0, xs_off, KSTEP, 128, HDIM, MROWS);
    tdm_load_2d(Wg + k0, ws_off, KSTEP, 64,  HDIM, HDIM);
  }
#else
  (void)xs_off; (void)ws_off;
  for (int c = tid; c < 128 * 8; c += 256) {           // X tile: 16B chunks
    int r = c >> 3, cc = (c & 7) * 8;
    *(v8bf*)(xs + r * LDPAD + cc) = *(const v8bf*)(Xg + (size_t)r * HDIM + k0 + cc);
  }
  for (int c = tid; c < 64 * 8; c += 256) {            // W tile
    int r = c >> 3, cc = (c & 7) * 8;
    *(v8bf*)(ws + r * LDPAD + cc) = *(const v8bf*)(Wg + (size_t)r * HDIM + k0 + cc);
  }
#endif
}

template <int PENDING>
static __device__ __forceinline__ void wait_stage(int tid) {
#if USE_TDM
  if ((tid >> 5) == 0) {
#if __has_builtin(__builtin_amdgcn_s_wait_tensorcnt)
    __builtin_amdgcn_s_wait_tensorcnt(PENDING);
#endif
  }
#endif
  __syncthreads();
}

__global__ __launch_bounds__(256) void qkv_gemm(
    const __bf16* __restrict__ X, const __bf16* __restrict__ W,
    const float* __restrict__ bias, __bf16* __restrict__ Y, int vmode) {
  __shared__ __align__(16) __bf16 lds[2 * XTILE_E + 2 * WTILE_E];  // 54KB

  int tid  = threadIdx.x;
  int lane = tid & 31;
  int wave = tid >> 5;
  int wy = wave >> 1;                 // 0..3 -> M sub-tile
  int wx = wave & 1;                  // 0..1 -> N sub-tile
  int m0 = (blockIdx.x >> 4) * 128;   // 64 M macro-tiles
  int n0 = (blockIdx.x & 15) * 64;    // 16 N macro-tiles

  const __bf16* Xg = X + (size_t)m0 * HDIM;
  const __bf16* Wg = W + (size_t)n0 * HDIM;

  v8f acc00 = vzero8(), acc01 = vzero8(), acc10 = vzero8(), acc11 = vzero8();

  // Capture the LDS symbol so GlobalOpt cannot fold its (TDM-written) loads.
  lds_escape(lds);

  // NOTE: no pointer/offset arrays derived from `lds` (they become
  // addrspacecast static initializers that ld.lld rejects) — compute
  // element offsets arithmetically per buffer index instead.
  fetch_stage(Xg, Wg, 0, lds, lds + 2 * XTILE_E,
              0u, (unsigned)(4 * XTILE_E), tid);

  for (int s = 0; s < NSTAGE; ++s) {
    int buf = s & 1;
    int nxt = buf ^ 1;
    if (s + 1 < NSTAGE) {
      fetch_stage(Xg, Wg, (s + 1) * KSTEP,
                  lds + nxt * XTILE_E,
                  lds + 2 * XTILE_E + nxt * WTILE_E,
                  (unsigned)(nxt * 2 * XTILE_E),            // byte offsets
                  (unsigned)(4 * XTILE_E + nxt * 2 * WTILE_E),
                  tid);
      wait_stage<2>(tid);   // stage s complete (TDM is in-order), s+1 in flight
    } else {
      wait_stage<0>(tid);
    }
    lds_escape(lds);        // TDM stores for stage s are now visible: reload LDS

    const __bf16* xb = lds + buf * XTILE_E + (size_t)wy * 32 * LDPAD;
    const __bf16* wb = lds + 2 * XTILE_E + buf * WTILE_E + (size_t)wx * 32 * LDPAD;
    for (int kk = 0; kk < KSTEP; kk += 32) {
      v16bf a0 = load_a_frag(xb,               LDPAD, lane, kk);
      v16bf a1 = load_a_frag(xb + 16 * LDPAD,  LDPAD, lane, kk);
      v16bf b0 = load_b_frag(wb,               LDPAD, lane, kk);
      v16bf b1 = load_b_frag(wb + 16 * LDPAD,  LDPAD, lane, kk);
      acc00 = wmma_bf16(a0, b0, acc00);
      acc01 = wmma_bf16(a0, b1, acc01);
      acc10 = wmma_bf16(a1, b0, acc10);
      acc11 = wmma_bf16(a1, b1, acc11);
    }
    lds_escape(lds);        // keep LDS reads above the refill barrier
    __syncthreads();        // everyone done reading buf before it is refilled
  }

  int gm = m0 + wy * 32, gn = n0 + wx * 32;
  store_tile_bf16(Y, bias, acc00, gm,      gn,      vmode, lane);
  store_tile_bf16(Y, bias, acc01, gm,      gn + 16, vmode, lane);
  store_tile_bf16(Y, bias, acc10, gm + 16, gn,      vmode, lane);
  store_tile_bf16(Y, bias, acc11, gm + 16, gn + 16, vmode, lane);
}

// ---- kernel 3: attention --------------------------------------------------
// One wave per (batch, head, 16-query block). LDS: 64KB f32 scores +
// 32KB bf16 probs = 96KB dynamic.

__global__ __launch_bounds__(32) void attn_kernel(
    const __bf16* __restrict__ Q, const __bf16* __restrict__ K,
    const __bf16* __restrict__ Vt, const float* __restrict__ mask,
    float* __restrict__ out) {
  extern __shared__ __align__(16) char smem[];
  float*  scores = (float*)smem;             // [16][SEQ]
  __bf16* probs  = (__bf16*)(smem + 65536);  // [16][SEQ]

  int lane = threadIdx.x & 31;
  int unit = blockIdx.x;             // 0 .. NB*NH*(SEQ/16)-1
  int qblk = unit & (SEQ / 16 - 1);
  int h    = (unit >> 6) & (NH - 1);
  int b    = unit >> 10;

  size_t head = (size_t)(b * NH + h) * (SEQ * DH);
  const __bf16* Qb = Q  + head + (size_t)qblk * 16 * DH;   // 16 x 64, ld=DH
  const __bf16* Kb = K  + head;                             // SEQ x 64, ld=DH
  const __bf16* Vb = Vt + head;                             // 64 x SEQ, ld=SEQ

  int col  = lane & 15;
  int mofs = (lane >> 4) << 3;

  // scores = Q K^T * (1/sqrt(d)) + mask
  v16bf aq0 = load_a_frag(Qb, DH, lane, 0);
  v16bf aq1 = load_a_frag(Qb, DH, lane, 32);
  for (int nt = 0; nt < SEQ / 16; ++nt) {
    v16bf kb0 = load_b_frag(Kb + (size_t)nt * 16 * DH, DH, lane, 0);
    v16bf kb1 = load_b_frag(Kb + (size_t)nt * 16 * DH, DH, lane, 32);
    v8f acc = vzero8();
    acc = wmma_bf16(aq0, kb0, acc);
    acc = wmma_bf16(aq1, kb1, acc);
    int c = nt * 16 + col;
    float mv = mask[b * SEQ + c];
    for (int r = 0; r < 8; ++r)
      scores[(mofs + r) * SEQ + c] = acc[r] * 0.125f + mv;
  }

  // softmax: lanes (l, l^16) share row l&15, each owns half the columns
  int row   = lane & 15;
  int cbase = (lane >> 4) * 512;
  float mx = -3.0e38f;
  for (int i = 0; i < 512; ++i) {
    int c = cbase + ((i + (row << 5)) & 511);   // stagger vs LDS banks
    mx = fmaxf(mx, scores[row * SEQ + c]);
  }
  mx = fmaxf(mx, __shfl_xor(mx, 16, 32));
  float sum = 0.0f;
  for (int i = 0; i < 512; ++i) {
    int c = cbase + ((i + (row << 5)) & 511);
    sum += __expf(scores[row * SEQ + c] - mx);
  }
  sum += __shfl_xor(sum, 16, 32);
  float inv = 1.0f / sum;
  for (int i = 0; i < 512; ++i) {
    int c = cbase + ((i + (row << 5)) & 511);
    probs[row * SEQ + c] = f2bf(__expf(scores[row * SEQ + c] - mx) * inv);
  }

  // ctx = P V   (K-dim = SEQ, N-dim = DH, V pre-transposed)
  v8f cacc[4];
  for (int t = 0; t < 4; ++t) cacc[t] = vzero8();
  for (int k0 = 0; k0 < SEQ; k0 += 32) {
    v16bf ap = load_a_frag(probs, SEQ, lane, k0);
    for (int t = 0; t < 4; ++t) {
      v16bf bv = load_b_frag(Vb + (size_t)t * 16 * SEQ, SEQ, lane, k0);
      cacc[t] = wmma_bf16(ap, bv, cacc[t]);
    }
  }
  for (int t = 0; t < 4; ++t) {
    int f = h * DH + t * 16 + col;
    for (int r = 0; r < 8; ++r) {
      int s = qblk * 16 + mofs + r;
      out[((size_t)b * SEQ + s) * HDIM + f] = cacc[t][r];
    }
  }
}

// ---- host-side launch ------------------------------------------------------

extern "C" void kernel_launch(void* const* d_in, const int* in_sizes, int n_in,
                              void* d_out, int out_size, void* d_ws,
                              size_t ws_size, hipStream_t stream) {
  (void)in_sizes; (void)n_in; (void)out_size; (void)ws_size;
  const float* act  = (const float*)d_in[0];
  const float* mask = (const float*)d_in[1];
  const float* qw   = (const float*)d_in[2];
  const float* qbi  = (const float*)d_in[3];
  const float* kw   = (const float*)d_in[4];
  const float* kbi  = (const float*)d_in[5];
  const float* vw   = (const float*)d_in[6];
  const float* vbi  = (const float*)d_in[7];
  float* out = (float*)d_out;

  // workspace layout (bytes): xb 16M | qwb/kwb/vwb 3*2M | Q/K/Vt 3*16M = 70M
  char* ws = (char*)d_ws;
  __bf16* xb  = (__bf16*)(ws);
  __bf16* qwb = (__bf16*)(ws + (16u << 20));
  __bf16* kwb = (__bf16*)(ws + (16u << 20) + (2u << 20));
  __bf16* vwb = (__bf16*)(ws + (16u << 20) + (4u << 20));
  __bf16* Qb  = (__bf16*)(ws + (16u << 20) + (6u << 20));
  __bf16* Kb  = Qb + (size_t)MROWS * HDIM;
  __bf16* Vtb = Kb + (size_t)MROWS * HDIM;

  cvt_bf16<<<2048, 256, 0, stream>>>(act, xb, MROWS * HDIM);
  cvt_bf16<<<1024, 256, 0, stream>>>(qw, qwb, HDIM * HDIM);
  cvt_bf16<<<1024, 256, 0, stream>>>(kw, kwb, HDIM * HDIM);
  cvt_bf16<<<1024, 256, 0, stream>>>(vw, vwb, HDIM * HDIM);

  // 64 x 16 macro-tiles of 128x64 -> 1024 blocks of 256 threads
  qkv_gemm<<<1024, 256, 0, stream>>>(xb, qwb, qbi, Qb, 0);
  qkv_gemm<<<1024, 256, 0, stream>>>(xb, kwb, kbi, Kb, 0);
  qkv_gemm<<<1024, 256, 0, stream>>>(xb, vwb, vbi, Vtb, 1);

  // one wave per (b,h,qblock): 8*16*64 = 8192 blocks, 96KB dynamic LDS
  attn_kernel<<<8192, 32, 98304, stream>>>(Qb, Kb, Vtb, mask, out);
}
